// ResidualVectorQuantizer_876173328853
// MI455X (gfx1250) — matrix operand order repaired and checked by reading, not compiled, not run on hardware
//
#include <hip/hip_runtime.h>
#include <hip/hip_bf16.h>

typedef __attribute__((ext_vector_type(16))) _Float16 v16h;
typedef __attribute__((ext_vector_type(8)))  float    v8f;

#define N_ROWS        65536
#define DIM           128
#define K_CB          1024
#define NSTAGE        4
#define ROWS_PER_BLK  64          // 4 waves * 16 rows
#define WAVES_PER_BLK 4
#define LDS_STRIDE    132         // 128 + 4 pad -> conflict-free row access
#define FRAG_PER_STAGE (64*4*32*16)   // 131072 halves per stage

// ---------------------------------------------------------------------------
// Prep: per codeword compute 1/||c||, emit normalized codebook pre-swizzled
// into a WMMA B-fragment layout in contiguous k order (matching A) (32xK chunk x 16 cols):
//   lane = ((K>>4)&1)*16 + (k&15), half h = K&15, chunk c = K>>5, tile t = k>>4
// ---------------------------------------------------------------------------
__global__ __launch_bounds__(256)
void rvq_prep(const float* __restrict__ codebooks,   // [4][1024][128]
              _Float16* __restrict__ fragB,          // [4][64][4][32][16]
              float* __restrict__ invnorm)           // [4096]
{
    int wave = threadIdx.x >> 5;
    int lane = threadIdx.x & 31;
    int g = blockIdx.x * 8 + wave;                   // codeword id 0..4095
    int s = g >> 10;
    int k = g & 1023;

    const float4 v = ((const float4*)(codebooks + (size_t)g * DIM))[lane];
    float ss = v.x*v.x + v.y*v.y + v.z*v.z + v.w*v.w;
    #pragma unroll
    for (int off = 16; off; off >>= 1) ss += __shfl_xor(ss, off, 32);
    float inv = 1.0f / fmaxf(sqrtf(ss), 1e-12f);
    if (lane == 0) invnorm[g] = inv;

    int t  = k >> 4;
    int kl = k & 15;
    float e[4] = {v.x, v.y, v.z, v.w};
    #pragma unroll
    for (int j = 0; j < 4; ++j) {
        int K = lane * 4 + j;
        int c = K >> 5;
        int p = (K >> 4) & 1;
        int h = K & 15;
        int laneF = p * 16 + kl;
        int addr = ((((s * 64 + t) * 4 + c) * 32 + laneF) * 16 + h);
        fragB[addr] = (_Float16)(e[j] * inv);
    }
}

// ---------------------------------------------------------------------------
// Main: each wave owns 16 rows. Per stage: build f16 A-fragments from the f32
// residual (LDS), sweep all 1024 codewords with v_wmma_f32_16x16x32_f16.
// Argmax trick: the 10 low mantissa bits of the f32 similarity are below the
// f16-input noise floor, so we overwrite them with the column index and track
// the running max with a single v_max_f32 per accumulator slot.
// The projection update / loss are recomputed in full f32.
// ---------------------------------------------------------------------------
__global__ __launch_bounds__(128)
void rvq_main(const float* __restrict__ x,
              const float* __restrict__ codebooks,
              const _Float16* __restrict__ fragB,
              const float* __restrict__ invnorm,
              float* __restrict__ xq_out,            // [65536][128]
              float* __restrict__ loss_part,         // [4096] per-wave partials
              float* __restrict__ idx_out)           // [65536][4] as float
{
    __shared__ float res[ROWS_PER_BLK * LDS_STRIDE];
    __shared__ int   kidx[WAVES_PER_BLK][16];

    const int wave = threadIdx.x >> 5;
    const int lane = threadIdx.x & 31;
    const int rw   = wave * 16;                      // first local row of wave
    const int rowBlock = blockIdx.x * ROWS_PER_BLK;
    const int part = lane >> 4;                      // 0/1 half of wave
    const int rloc = lane & 15;                      // row-in-strip / col-in-tile

    // ---- load x -> residual in LDS (coalesced 512B per wave access) ----
    #pragma unroll
    for (int it = 0; it < 4; ++it) {
        int r = rw + it * 4 + (lane >> 3);
        const float4* src = (const float4*)(x + (size_t)(rowBlock + r) * DIM);
        #pragma unroll
        for (int cc = 0; cc < 4; ++cc) {
            int c4 = cc * 8 + (lane & 7);
            *(float4*)&res[r * LDS_STRIDE + c4 * 4] = src[c4];
        }
    }

    float lossAcc = 0.0f;
    const float* rrow = &res[(rw + rloc) * LDS_STRIDE];

    for (int s = 0; s < NSTAGE; ++s) {
        // ---- build A fragments (16x32 f16 per chunk), reused for all tiles ----
        v16h a[4];
        #pragma unroll
        for (int c = 0; c < 4; ++c) {
            int base = c * 32 + part * 16;
            #pragma unroll
            for (int h = 0; h < 8; ++h) {
                a[c][h]     = (_Float16)rrow[base + h];
                a[c][h + 8] = (_Float16)rrow[base + 8 + h];
            }
        }

        // ---- argmax sweep over 1024 codewords (64 tiles of 16) ----
        float bv[8];
        #pragma unroll
        for (int j = 0; j < 8; ++j) bv[j] = -3.4e38f;

        const _Float16* fB = fragB + s * FRAG_PER_STAGE;
        #pragma unroll 2
        for (int t = 0; t < 64; ++t) {
            if ((t & 1) == 0)   // prefetch ~16KB (4 tiles) ahead (tail overshoot stays in ws)
                __builtin_prefetch(fB + (t * 4 + 16) * 32 * 16, 0, 1);
            v8f acc = {0.f, 0.f, 0.f, 0.f, 0.f, 0.f, 0.f, 0.f};
            #pragma unroll
            for (int c = 0; c < 4; ++c) {
                v16h b = *(const v16h*)(fB + (((t * 4 + c) * 32 + lane) * 16));
                acc = __builtin_amdgcn_wmma_f32_16x16x32_f16(
                          false, a[c], false, b, (short)0, acc, false, false);
            }
            unsigned col = (unsigned)(t * 16 + rloc);
            #pragma unroll
            for (int j = 0; j < 8; ++j) {
                float pv = __uint_as_float(
                    (__float_as_uint(acc[j]) & 0xFFFFFC00u) | col);
                bv[j] = fmaxf(bv[j], pv);
            }
        }

        // ---- max-reduce packed (value|index) across the 16 lanes of each half ----
        #pragma unroll
        for (int j = 0; j < 8; ++j) {
            #pragma unroll
            for (int off = 8; off; off >>= 1)
                bv[j] = fmaxf(bv[j], __shfl_xor(bv[j], off, 32));
        }
        int bi[8];
        #pragma unroll
        for (int j = 0; j < 8; ++j) bi[j] = (int)(__float_as_uint(bv[j]) & 0x3FFu);

        if (lane == 0) {
            #pragma unroll
            for (int j = 0; j < 8; ++j) kidx[wave][j] = bi[j];
        }
        if (lane == 16) {
            #pragma unroll
            for (int j = 0; j < 8; ++j) kidx[wave][8 + j] = bi[j];
        }
        if (lane == 0 || lane == 16) {
            int rbase = rowBlock + rw + (lane == 16 ? 8 : 0);
            #pragma unroll
            for (int j = 0; j < 8; ++j)
                idx_out[(size_t)(rbase + j) * NSTAGE + s] = (float)bi[j];
        }
        __syncthreads();

        // ---- f32 projection update: proj = (res . c_n) c_n; loss in f32 ----
        int   krow = kidx[wave][rloc];
        float inv  = invnorm[s * K_CB + krow];
        const float* cpart = codebooks + ((size_t)s * K_CB + krow) * DIM + part * 64;
        float* rr = &res[(rw + rloc) * LDS_STRIDE + part * 64];

        float tdot = 0.0f;
        #pragma unroll 4
        for (int i = 0; i < 16; ++i) {
            float4 r4 = *(float4*)&rr[i * 4];
            float4 c4 = ((const float4*)cpart)[i];
            tdot += r4.x * c4.x + r4.y * c4.y + r4.z * c4.z + r4.w * c4.w;
            float dx = c4.x - r4.x, dy = c4.y - r4.y, dz = c4.z - r4.z, dw = c4.w - r4.w;
            lossAcc += dx * dx + dy * dy + dz * dz + dw * dw;
        }
        tdot += __shfl_xor(tdot, 16, 32);            // combine the two halves
        float coef = tdot * inv * inv;               // (res.c_n)/||c|| applied to raw c
        #pragma unroll 4
        for (int i = 0; i < 16; ++i) {
            float4 r4 = *(float4*)&rr[i * 4];
            float4 c4 = ((const float4*)cpart)[i];
            r4.x -= coef * c4.x; r4.y -= coef * c4.y;
            r4.z -= coef * c4.z; r4.w -= coef * c4.w;
            *(float4*)&rr[i * 4] = r4;
        }
        __syncthreads();
    }

    // ---- x_q = x - residual_final ----
    #pragma unroll
    for (int it = 0; it < 4; ++it) {
        int r = rw + it * 4 + (lane >> 3);
        size_t grow = (size_t)(rowBlock + r) * DIM;
        const float4* src = (const float4*)(x + grow);
        float4*       dst = (float4*)(xq_out + grow);
        #pragma unroll
        for (int cc = 0; cc < 4; ++cc) {
            int c4 = cc * 8 + (lane & 7);
            float4 xv = src[c4];
            float4 rv = *(float4*)&res[r * LDS_STRIDE + c4 * 4];
            dst[c4] = make_float4(xv.x - rv.x, xv.y - rv.y, xv.z - rv.z, xv.w - rv.w);
        }
    }

    // ---- deterministic loss partial: one value per wave ----
    #pragma unroll
    for (int off = 16; off; off >>= 1) lossAcc += __shfl_xor(lossAcc, off, 32);
    if (lane == 0) loss_part[blockIdx.x * WAVES_PER_BLK + wave] = lossAcc;
}

// ---------------------------------------------------------------------------
// Deterministic final loss reduction: mean over stages of 1.25 * mean(sqdiff)
// ---------------------------------------------------------------------------
__global__ __launch_bounds__(256)
void rvq_loss_final(const float* __restrict__ loss_part, float* __restrict__ out_loss)
{
    __shared__ float sm[256];
    float v = 0.0f;
    for (int i = threadIdx.x; i < 4096; i += 256) v += loss_part[i];
    sm[threadIdx.x] = v;
    __syncthreads();
    for (int st = 128; st; st >>= 1) {
        if (threadIdx.x < st) sm[threadIdx.x] += sm[threadIdx.x + st];
        __syncthreads();
    }
    if (threadIdx.x == 0)
        *out_loss = sm[0] * (1.25f / ((float)NSTAGE * (float)N_ROWS * (float)DIM));
}

extern "C" void kernel_launch(void* const* d_in, const int* in_sizes, int n_in,
                              void* d_out, int out_size, void* d_ws, size_t ws_size,
                              hipStream_t stream)
{
    (void)in_sizes; (void)n_in; (void)out_size; (void)ws_size;
    const float* x  = (const float*)d_in[0];          // [65536,128]
    const float* cb = (const float*)d_in[1];          // [4,1024,128]

    float* out   = (float*)d_out;
    float* xq    = out;                               // 8388608 floats
    float* loss  = out + (size_t)N_ROWS * DIM;        // 1 float
    float* idxo  = loss + 1;                          // 262144 floats

    char* ws = (char*)d_ws;
    _Float16* fragB = (_Float16*)ws;                          // 1 MB
    float* invn  = (float*)(ws + 4 * FRAG_PER_STAGE * 2);     // 16 KB
    float* lpart = invn + 4096;                               // 16 KB

    rvq_prep<<<512, 256, 0, stream>>>(cb, fragB, invn);
    rvq_main<<<N_ROWS / ROWS_PER_BLK, 128, 0, stream>>>(x, cb, fragB, invn,
                                                        xq, lpart, idxo);
    rvq_loss_final<<<1, 256, 0, stream>>>(lpart, loss);
}
